// HistoricalDecoder_5377299054978
// MI455X (gfx1250) — matrix operand order repaired and checked by reading, not compiled, not run on hardware
//
#include <hip/hip_runtime.h>

// ---------------------------------------------------------------------------
// HistoricalDecoder fused forward for gfx1250 (MI455X)
//   h = relu([src|dst] @ W1 + b1);  out = [h | hist[key]] @ W2 + b2
// fc1 via bf16-split WMMA (hi*hi + hi*lo + lo*hi) ~= fp32 accuracy.
// W1 pre-swizzled to bf16 fragment layout; per-k-tile slices staged in LDS
// with double-buffered global_load_async_to_lds_b128 (ASYNCcnt).
// ---------------------------------------------------------------------------

#define BATCH_N 65536
#define KDIM    512
#define HID     256
#define BM      128     // batch rows per workgroup (8 waves x 16 rows)
#define WROWS   16      // rows per wave
#define SLICE   8192    // ushort elements per k-tile slice of W1 (32*256)

typedef __attribute__((ext_vector_type(16))) __bf16         v16bf;
typedef __attribute__((ext_vector_type(8)))  float          v8f;
typedef __attribute__((ext_vector_type(16))) unsigned short v16u;

static __device__ __forceinline__ unsigned short f2bf_bits(float f) {
  union { float f; unsigned u; } x; x.f = f;
  unsigned u = x.u + 0x7FFFu + ((x.u >> 16) & 1u);   // RNE
  return (unsigned short)(u >> 16);
}
static __device__ __forceinline__ float bfbits2f(unsigned short h) {
  union { unsigned u; float f; } x; x.u = ((unsigned)h) << 16;
  return x.f;
}

// Pre-swizzle W1 [512 x 256] fp32 into bf16 hi/lo arrays laid out exactly as
// the per-lane B-matrix fragments this code feeds V_WMMA_F32_16X16X32_BF16 (contiguous k order):
//   flat idx = (((kt*16 + nt)*32 + lane)*16 + e)
//   k = kt*32 + (lane/16)*16 + e ; n = nt*16 + (lane%16)
// -> each lane's 16-element fragment is 32 contiguous bytes.
__global__ void prep_w1_kernel(const float* __restrict__ W1,
                               unsigned short* __restrict__ whi,
                               unsigned short* __restrict__ wlo) {
  int idx  = blockIdx.x * blockDim.x + threadIdx.x;   // 131072 total
  int e    = idx & 15;
  int lane = (idx >> 4) & 31;
  int nt   = (idx >> 9) & 15;
  int kt   = (idx >> 13) & 15;
  int k = kt * 32 + (lane >> 4) * 16 + e;
  int n = nt * 16 + (lane & 15);
  float f = W1[(size_t)k * HID + n];
  unsigned short h = f2bf_bits(f);
  whi[idx] = h;
  wlo[idx] = f2bf_bits(f - bfbits2f(h));
}

// Issue 4 async 16B copies (one k-tile slice is 1024 chunks / 256 threads).
static __device__ __forceinline__ void async_copy_slice(
    const unsigned short* __restrict__ gsrc, unsigned lds_byte_base, int tid) {
  #pragma unroll
  for (int i = 0; i < 4; ++i) {
    int c = tid + 256 * i;                               // 16B chunk id
    unsigned long long ga =
        (unsigned long long)(uintptr_t)(gsrc + (size_t)c * 8);
    unsigned la = lds_byte_base + (unsigned)c * 16u;
    asm volatile("global_load_async_to_lds_b128 %0, %1, off"
                 :: "v"(la), "v"(ga) : "memory");
  }
}

__global__ __launch_bounds__(256) void fused_fwd_kernel(
    const float* __restrict__ src, const float* __restrict__ dst,
    const int* __restrict__ key_idx, const float* __restrict__ hist,
    const unsigned short* __restrict__ whi, const unsigned short* __restrict__ wlo,
    const float* __restrict__ b1, const float* __restrict__ W2,
    const float* __restrict__ b2, float* __restrict__ out) {
  __shared__ alignas(16) unsigned short bufH[2][SLICE];  // 32 KB
  __shared__ alignas(16) unsigned short bufL[2][SLICE];  // 32 KB
  __shared__ float rowsum[BM];

  const int tid  = threadIdx.x;
  const int lane = tid & 31;
  const int w    = tid >> 5;
  if (tid < BM) rowsum[tid] = 0.0f;

  const unsigned ldsH0 = (unsigned)(uintptr_t)&bufH[0][0];
  const unsigned ldsH1 = (unsigned)(uintptr_t)&bufH[1][0];
  const unsigned ldsL0 = (unsigned)(uintptr_t)&bufL[0][0];
  const unsigned ldsL1 = (unsigned)(uintptr_t)&bufL[1][0];

  // Prefetch k-tiles 0 and 1.
  async_copy_slice(whi + 0 * SLICE, ldsH0, tid);
  async_copy_slice(wlo + 0 * SLICE, ldsL0, tid);
  async_copy_slice(whi + 1 * SLICE, ldsH1, tid);
  async_copy_slice(wlo + 1 * SLICE, ldsL1, tid);

  const int g = lane >> 4;                                    // lane half
  const int mrow = blockIdx.x * BM + w * WROWS + (lane & 15); // A-matrix row
  const float* srow = src + (size_t)mrow * 256;
  const float* drow = dst + (size_t)mrow * 256;

  const v8f vzero = {0.f, 0.f, 0.f, 0.f, 0.f, 0.f, 0.f, 0.f};
  v8f acc[16];
  #pragma unroll
  for (int i = 0; i < 16; ++i) acc[i] = vzero;

  #pragma unroll 1
  for (int kt = 0; kt < 16; ++kt) {
    // A fragment (16-bit 16x32 layout): lane half g needs k = g*16..g*16+15
    // within this 32-wide k-tile (matches contiguous W1 fragment order).
    const float* base = (kt < 8) ? (srow + kt * 32) : (drow + (kt - 8) * 32);
    const float4* p0 = (const float4*)(base + g * 16);
    float4 x0 = p0[0], x1 = p0[1], x2 = p0[2], x3 = p0[3];
    float rr[16] = {x0.x, x0.y, x0.z, x0.w, x1.x, x1.y, x1.z, x1.w,
                    x2.x, x2.y, x2.z, x2.w, x3.x, x3.y, x3.z, x3.w};
    v16bf ahi, alo;
    #pragma unroll
    for (int i = 0; i < 16; ++i) {
      float f = rr[i];
      __bf16 h = (__bf16)f;              // hardware cvt, RNE
      ahi[i] = h;
      alo[i] = (__bf16)(f - (float)h);
    }

    // Wait for this tile's async copies (next tile may stay in flight),
    // then sync so every wave's copies are visible.
    if (kt < 15) asm volatile("s_wait_asynccnt 8" ::: "memory");
    else         asm volatile("s_wait_asynccnt 0" ::: "memory");
    __syncthreads();

    const unsigned short* sh = &bufH[kt & 1][0];
    const unsigned short* sl = &bufL[kt & 1][0];
    #pragma unroll
    for (int nt = 0; nt < 16; ++nt) {
      size_t off = ((size_t)nt * 32 + lane) * 16;
      v16bf bhi = __builtin_bit_cast(v16bf, *(const v16u*)(sh + off));
      v16bf blo = __builtin_bit_cast(v16bf, *(const v16u*)(sl + off));
      acc[nt] = __builtin_amdgcn_wmma_f32_16x16x32_bf16(
          false, ahi, false, bhi, (short)0, acc[nt], false, false);
      acc[nt] = __builtin_amdgcn_wmma_f32_16x16x32_bf16(
          false, alo, false, bhi, (short)0, acc[nt], false, false);
      acc[nt] = __builtin_amdgcn_wmma_f32_16x16x32_bf16(
          false, ahi, false, blo, (short)0, acc[nt], false, false);
    }
    __syncthreads();   // all waves done reading buf[kt&1]

    if (kt + 2 < 16) {
      unsigned dH = (kt & 1) ? ldsH1 : ldsH0;
      unsigned dL = (kt & 1) ? ldsL1 : ldsL0;
      async_copy_slice(whi + (size_t)(kt + 2) * SLICE, dH, tid);
      async_copy_slice(wlo + (size_t)(kt + 2) * SLICE, dL, tid);
    }
  }

  // Epilogue: bias + relu + fc2 first half, reduced per output row.
  // C layout: lane -> n = nt*16 + (lane%16); VGPR r -> m = r + 8*g.
  float sr[8] = {0.f, 0.f, 0.f, 0.f, 0.f, 0.f, 0.f, 0.f};
  #pragma unroll
  for (int nt = 0; nt < 16; ++nt) {
    int n = nt * 16 + (lane & 15);
    float bias = b1[n];
    float w2v  = W2[n];
    #pragma unroll
    for (int r = 0; r < 8; ++r) {
      float h = acc[nt][r] + bias;
      h = fmaxf(h, 0.0f);
      sr[r] += h * w2v;
    }
  }
  #pragma unroll
  for (int r = 0; r < 8; ++r)
    atomicAdd(&rowsum[w * WROWS + g * 8 + r], sr[r]);
  __syncthreads();

  // Historical gather + fc2 second half: 2 threads per row, 512B each.
  {
    int row  = tid >> 1;
    int half = tid & 1;
    int b    = blockIdx.x * BM + row;
    int key  = key_idx[b];
    const float4* h4 = (const float4*)(hist + (size_t)key * HID + half * 128);
    const float4* w4 = (const float4*)(W2 + HID + half * 128);
    float s = 0.f;
    #pragma unroll
    for (int j = 0; j < 32; ++j) {
      float4 a = h4[j], v = w4[j];
      s += a.x * v.x + a.y * v.y + a.z * v.z + a.w * v.w;
    }
    atomicAdd(&rowsum[row], s);
  }
  __syncthreads();

  if (tid < BM)
    out[blockIdx.x * BM + tid] = rowsum[tid] + b2[0];
}

extern "C" void kernel_launch(void* const* d_in, const int* in_sizes, int n_in,
                              void* d_out, int out_size, void* d_ws, size_t ws_size,
                              hipStream_t stream) {
  const float* src  = (const float*)d_in[0];
  const float* dst  = (const float*)d_in[1];
  const int*   key  = (const int*)d_in[2];
  const float* hist = (const float*)d_in[3];
  const float* W1   = (const float*)d_in[4];
  const float* b1   = (const float*)d_in[5];
  const float* W2   = (const float*)d_in[6];
  const float* b2   = (const float*)d_in[7];
  float* out = (float*)d_out;

  unsigned short* whi = (unsigned short*)d_ws;            // 512*256 bf16 hi
  unsigned short* wlo = whi + (size_t)KDIM * HID;         // 512*256 bf16 lo

  prep_w1_kernel<<<(KDIM * HID) / 256, 256, 0, stream>>>(W1, whi, wlo);
  fused_fwd_kernel<<<BATCH_N / BM, 256, 0, stream>>>(
      src, dst, key, hist, whi, wlo, b1, W2, b2, out);
}
